// LSTMAutoEncoder_55825984913507
// MI455X (gfx1250) — matrix operand order repaired and checked by reading, not compiled, not run on hardware
//
#include <hip/hip_runtime.h>
#include <hip/hip_bf16.h>

// ---------------------------------------------------------------------------
// LSTM autoencoder, B=512, L=360, H=512, 360 decoder steps.
// bf16 WMMA (v_wmma_f32_16x16x32_bf16), f32 accumulate, wave32.
// TDM (tensor_load_to_lds) stages the shared h-strip per block into LDS.
// ---------------------------------------------------------------------------

typedef __bf16 bf16_t;
typedef __attribute__((ext_vector_type(16))) __bf16 v16bf;
typedef __attribute__((ext_vector_type(8)))  __bf16 v8bf;
typedef __attribute__((ext_vector_type(8)))  float  v8f;

typedef unsigned int v4ui __attribute__((ext_vector_type(4)));
typedef int          v8si __attribute__((ext_vector_type(8)));
typedef int          v4si __attribute__((ext_vector_type(4)));

#define BATCH 512
#define SEQL  360
#define HID   512
#define G4H   2048   // 4*H
#define LPAD  384    // K-pad of L (360 -> 384, multiple of 32)
#define NPAD  368    // N-pad of fc output (360 -> 368, multiple of 16)

union V16U { v16bf v; v8bf h[2]; };

static __device__ inline v8f vzero8f() {
    v8f z = {0.f,0.f,0.f,0.f,0.f,0.f,0.f,0.f};
    return z;
}

// round-to-nearest-even f32 -> bf16
static __device__ inline bf16_t f2bf(float f) {
    union { float f; unsigned u; } a; a.f = f;
    unsigned r = a.u + 0x7FFFu + ((a.u >> 16) & 1u);
    union { unsigned short s; bf16_t b; } o; o.s = (unsigned short)(r >> 16);
    return o.b;
}

static __device__ inline float sigmoidf(float x) {
    return 1.0f / (1.0f + __expf(-x));
}
static __device__ inline float tanhfast(float x) {
    float e = __expf(2.0f * x);
    return 1.0f - 2.0f / (e + 1.0f);   // saturates correctly at +-1
}

// A fragment: 16x32 bf16 tile of row-major matrix, base -> element [row0][k0].
// ISA layout: lane l (r=l&15, hi=l>>4): K offsets {hi*8..+7, 16+hi*8..+7}.
static __device__ inline v16bf load_fragA(const bf16_t* base, int ld, int lane) {
    int r  = lane & 15;
    int hi = lane >> 4;
    const bf16_t* p = base + r * ld + hi * 8;
    V16U u;
    u.h[0] = *(const v8bf*)(p);
    u.h[1] = *(const v8bf*)(p + 16);
    return u.v;
}
// Same pattern reading the staged strip out of LDS.
static __device__ inline v16bf load_fragA_lds(const bf16_t* base, int ld, int lane) {
    int r  = lane & 15;
    int hi = lane >> 4;
    const bf16_t* p = base + r * ld + hi * 8;
    V16U u;
    u.h[0] = *(const v8bf*)(p);
    u.h[1] = *(const v8bf*)(p + 16);
    return u.v;
}

// B fragment: 32x16 (KxN) where B[k][n] = W[n][k], W row-major (N x K).
// Lane l: n=l&15, K offsets {hi*8..+7, 16+hi*8..+7} (matches A).
static __device__ inline v16bf load_fragB(const bf16_t* base, int ld, int lane) {
    int n  = lane & 15;
    int hi = lane >> 4;
    const bf16_t* p = base + n * ld + hi * 8;
    V16U u;
    u.h[0] = *(const v8bf*)(p);
    u.h[1] = *(const v8bf*)(p + 16);
    return u.v;
}

// ---------------------------------------------------------------------------
// TDM: stage `rows` x `rowlen` bf16 tile (row stride `stride` elems) from
// global into LDS at byte offset lds_off. D# per CDNA5 ISA §8.3/8.4.
// This toolchain exposes the 6-arg builtin:
//   (uint32x4 g0, int32x8 g1, int32x4 g2, int32x4 g3, int32x8 ext, i32 cpol)
// ---------------------------------------------------------------------------
#if __has_builtin(__builtin_amdgcn_tensor_load_to_lds)
#define HAVE_TDM 1
static __device__ inline void tdm_load_tile(unsigned lds_off, const void* gptr,
                                            unsigned rows, unsigned rowlen,
                                            unsigned stride) {
    unsigned long long ga = (unsigned long long)gptr;
    v4ui g0;
    g0[0] = 1u;                                        // count=1, user-mode
    g0[1] = lds_off;                                   // lds_addr [63:32]
    g0[2] = (unsigned)(ga & 0xFFFFFFFFu);              // global_addr [95:64]
    g0[3] = (unsigned)((ga >> 32) & 0x01FFFFFFu)       // global_addr [120:96]
          | 0x80000000u;                               // type=2  [127:126]
    v8si g1;
    g1[0] = (int)(1u << 16);                           // data_size=2B
    g1[1] = (int)((rowlen & 0xFFFFu) << 16);           // tensor_dim0 lo16
    g1[2] = (int)((rowlen >> 16) & 0xFFFFu)            // tensor_dim0 hi16
          | (int)((rows & 0xFFFFu) << 16);             // tensor_dim1 lo16
    g1[3] = (int)((rows >> 16) & 0xFFFFu)              // tensor_dim1 hi16
          | (int)((rowlen & 0xFFFFu) << 16);           // tile_dim0
    g1[4] = (int)(rows & 0xFFFFu);                     // tile_dim1 (tile_dim2=0)
    g1[5] = (int)stride;                               // tensor_dim0_stride lo32
    g1[6] = 0;                                         // strides hi bits
    g1[7] = 0;
    v4si z4 = {0, 0, 0, 0};
    v8si z8 = {0, 0, 0, 0, 0, 0, 0, 0};
    __builtin_amdgcn_tensor_load_to_lds(g0, g1, z4, z4, z8, 0);
}
#else
#define HAVE_TDM 0
#endif

// Stage h strip (16 rows x 512) for mtile `mt` into lds (all 256 threads call).
static __device__ inline void stage_h_strip(bf16_t* lds, const bf16_t* h_in, int mt) {
#if HAVE_TDM
    if (threadIdx.x < 32) {
        // Low 32 bits of the generic address of a __shared__ object are the
        // LDS byte address (flat aperture rule).
        unsigned lds_off = (unsigned)(unsigned long long)lds;
        tdm_load_tile(lds_off, h_in + (size_t)mt * 16 * HID,
                      16u, (unsigned)HID, (unsigned)HID);
        __builtin_amdgcn_s_wait_tensorcnt(0);
    }
    __syncthreads();
#else
    const int elems = 16 * HID;              // 8192 bf16 = 1024 x v8bf
    for (int i = threadIdx.x; i < elems / 8; i += 256) {
        ((v8bf*)lds)[i] = ((const v8bf*)(h_in + (size_t)mt * 16 * HID))[i];
    }
    __syncthreads();
#endif
}

// ---------------------------------------------------------------------------
// f32 -> bf16 convert with zero padding (rows and cols)
// ---------------------------------------------------------------------------
__global__ void __launch_bounds__(256)
convert_pad_kernel(const float* __restrict__ src, bf16_t* __restrict__ dst,
                   int srows, int scols, int drows, int dcols) {
    int idx = blockIdx.x * 256 + threadIdx.x;
    int total = drows * dcols;
    if (idx >= total) return;
    int r = idx / dcols, c = idx - r * dcols;
    float v = (r < srows && c < scols) ? src[r * scols + c] : 0.0f;
    dst[idx] = f2bf(v);
}

// ---------------------------------------------------------------------------
// Generic C(f32) = A(bf16, M x K) @ W(bf16, N x K)^T ; one wave per 16x16 tile
// ---------------------------------------------------------------------------
__global__ void __launch_bounds__(256)
gemm_bf16_kernel(const bf16_t* __restrict__ A, int lda,
                 const bf16_t* __restrict__ W, int ldw,
                 float* __restrict__ C, int ldc,
                 int ntiles, int ktiles) {
    int wave = (blockIdx.x * 256 + threadIdx.x) >> 5;
    int lane = threadIdx.x & 31;
    int mt = wave / ntiles;
    int nt = wave - mt * ntiles;
    v8f acc = vzero8f();
    const bf16_t* a0 = A + (size_t)(mt * 16) * lda;
    const bf16_t* b0 = W + (size_t)(nt * 16) * ldw;
    for (int kt = 0; kt < ktiles; ++kt) {
        v16bf a = load_fragA(a0 + kt * 32, lda, lane);
        v16bf b = load_fragB(b0 + kt * 32, ldw, lane);
        acc = __builtin_amdgcn_wmma_f32_16x16x32_bf16(
            false, a, false, b, (short)0, acc, false, false);
    }
    int row = mt * 16 + (lane >> 4) * 8;
    int col = nt * 16 + (lane & 15);
#pragma unroll
    for (int r = 0; r < 8; ++r)
        C[(size_t)(row + r) * ldc + col] = acc[r];
}

// ---------------------------------------------------------------------------
// xproj += bias1[col] + bias2[col]
// ---------------------------------------------------------------------------
__global__ void __launch_bounds__(256)
addbias2_kernel(float* __restrict__ xp, const float* __restrict__ b1,
                const float* __restrict__ b2, int rows, int cols) {
    int idx = blockIdx.x * 256 + threadIdx.x;
    if (idx >= rows * cols) return;
    int c = idx % cols;
    xp[idx] += b1[c] + b2[c];
}

// ---------------------------------------------------------------------------
// Encoder cell (h0 = c0 = 0): gates (512 x 2048) -> c (f32), h (bf16)
// ---------------------------------------------------------------------------
__global__ void __launch_bounds__(256)
enc_cell_kernel(const float* __restrict__ gates,
                const float* __restrict__ bih, const float* __restrict__ bhh,
                float* __restrict__ c, bf16_t* __restrict__ h) {
    int idx = blockIdx.x * 256 + threadIdx.x;
    if (idx >= BATCH * HID) return;
    int row = idx >> 9, col = idx & (HID - 1);
    const float* g = gates + (size_t)row * G4H;
    float gi = g[col]           + bih[col]           + bhh[col];
    float gg = g[2 * HID + col] + bih[2 * HID + col] + bhh[2 * HID + col];
    float go = g[3 * HID + col] + bih[3 * HID + col] + bhh[3 * HID + col];
    float iv = sigmoidf(gi), gv = tanhfast(gg), ov = sigmoidf(go);
    float cn = iv * gv;                 // f * c0 == 0
    c[idx] = cn;
    h[idx] = f2bf(ov * tanhfast(cn));
}

// ---------------------------------------------------------------------------
// Fused decoder step + previous-step fc projection.
//   Blocks [0,128):   gate blocks. block = (mt, jgroup). Waves: jt=jgroup*8+w.
//                     Each wave: 16x16 h-tile, accumulating all 4 gate tiles
//                     (cols j, j+H, j+2H, j+3H of dec_Whh) -> LSTM epilogue.
//   Blocks [128,224): pred blocks. block = (mt, ngroup). Waves: nt=ngroup*8+w.
//                     pred_{t-1} = h_in @ fc_W^T + fc_b -> out_prev.
// The shared 16x512 h_in strip of each block is TDM-staged into LDS.
// ---------------------------------------------------------------------------
__global__ void __launch_bounds__(256)
step_fused_kernel(const bf16_t* __restrict__ h_in, bf16_t* __restrict__ h_out,
                  const bf16_t* __restrict__ Whh,     // 2048 x 512 bf16
                  const float*  __restrict__ xprojb,  // 512 x 2048, biased
                  float* __restrict__ c,
                  const bf16_t* __restrict__ fcW,     // 368 x 512 bf16
                  const float*  __restrict__ fcb,
                  float* __restrict__ out_prev) {     // null on step 0
    __shared__ __align__(64) bf16_t lds_h[16 * HID];  // 16 KB
    int w    = threadIdx.x >> 5;
    int lane = threadIdx.x & 31;

    if (blockIdx.x < 128) {
        // ---------------- gate blocks ----------------
        int mt = blockIdx.x >> 2;          // 0..31
        int jg = blockIdx.x & 3;           // 0..3
        int jt = jg * 8 + w;               // 0..31
        stage_h_strip(lds_h, h_in, mt);

        v8f acc[4];
#pragma unroll
        for (int g = 0; g < 4; ++g) acc[g] = vzero8f();

#pragma unroll 4
        for (int kt = 0; kt < HID / 32; ++kt) {
            v16bf a = load_fragA_lds(lds_h + kt * 32, HID, lane);
#pragma unroll
            for (int g = 0; g < 4; ++g) {
                const bf16_t* b0 = Whh + (size_t)(g * HID + jt * 16) * HID + kt * 32;
                v16bf b = load_fragB(b0, HID, lane);
                acc[g] = __builtin_amdgcn_wmma_f32_16x16x32_bf16(
                    false, a, false, b, (short)0, acc[g], false, false);
            }
        }

        int row = mt * 16 + (lane >> 4) * 8;
        int col = jt * 16 + (lane & 15);
#pragma unroll
        for (int r = 0; r < 8; ++r) {
            int rr = row + r;
            const float* xp = xprojb + (size_t)rr * G4H;
            float gi = acc[0][r] + xp[col];
            float gf = acc[1][r] + xp[HID + col];
            float gg = acc[2][r] + xp[2 * HID + col];
            float go = acc[3][r] + xp[3 * HID + col];
            float iv = sigmoidf(gi), fv = sigmoidf(gf);
            float gv = tanhfast(gg), ov = sigmoidf(go);
            int cidx = rr * HID + col;
            float cn = fv * c[cidx] + iv * gv;
            c[cidx] = cn;
            h_out[cidx] = f2bf(ov * tanhfast(cn));
        }
    } else {
        // ---------------- pred blocks ----------------
        if (out_prev == nullptr) return;   // uniform: whole block exits
        int pb = blockIdx.x - 128;         // 0..95
        int mt = pb / 3;                   // 0..31
        int ng = pb - mt * 3;              // 0..2
        int nt = ng * 8 + w;               // 0..23 (23 is padding -> skip)
        stage_h_strip(lds_h, h_in, mt);
        if (nt >= 23) return;

        v8f acc = vzero8f();
        const bf16_t* b0 = fcW + (size_t)(nt * 16) * HID;
#pragma unroll 4
        for (int kt = 0; kt < HID / 32; ++kt) {
            v16bf a = load_fragA_lds(lds_h + kt * 32, HID, lane);
            v16bf b = load_fragB(b0 + kt * 32, HID, lane);
            acc = __builtin_amdgcn_wmma_f32_16x16x32_bf16(
                false, a, false, b, (short)0, acc, false, false);
        }
        int row = mt * 16 + (lane >> 4) * 8;
        int col = nt * 16 + (lane & 15);
        if (col < SEQL) {
            float bias = fcb[col];
#pragma unroll
            for (int r = 0; r < 8; ++r)
                out_prev[(size_t)(row + r) * (SEQL * SEQL) + col] = acc[r] + bias;
        }
    }
}

// ---------------------------------------------------------------------------
// Standalone pred (used once for the final step): h @ fc_W^T + fc_b
// ---------------------------------------------------------------------------
__global__ void __launch_bounds__(256)
pred_kernel(const bf16_t* __restrict__ h, const bf16_t* __restrict__ fcW,
            const float* __restrict__ fcb, float* __restrict__ out_t) {
    int wave = (blockIdx.x * 256 + threadIdx.x) >> 5;
    int lane = threadIdx.x & 31;
    int mt = wave / 23;
    int nt = wave - mt * 23;
    v8f acc = vzero8f();
    const bf16_t* a0 = h   + (size_t)(mt * 16) * HID;
    const bf16_t* b0 = fcW + (size_t)(nt * 16) * HID;
#pragma unroll 4
    for (int kt = 0; kt < HID / 32; ++kt) {
        v16bf a = load_fragA(a0 + kt * 32, HID, lane);
        v16bf b = load_fragB(b0 + kt * 32, HID, lane);
        acc = __builtin_amdgcn_wmma_f32_16x16x32_bf16(
            false, a, false, b, (short)0, acc, false, false);
    }
    int row = mt * 16 + (lane >> 4) * 8;
    int col = nt * 16 + (lane & 15);
    if (col < SEQL) {
        float bias = fcb[col];
#pragma unroll
        for (int r = 0; r < 8; ++r)
            out_t[(size_t)(row + r) * (SEQL * SEQL) + col] = acc[r] + bias;
    }
}

// ---------------------------------------------------------------------------
// Host side
// ---------------------------------------------------------------------------
extern "C" void kernel_launch(void* const* d_in, const int* in_sizes, int n_in,
                              void* d_out, int out_size, void* d_ws, size_t ws_size,
                              hipStream_t stream) {
    const float* x       = (const float*)d_in[0];   // 512 x 1 x 360
    const float* enc_Wih = (const float*)d_in[1];   // 2048 x 360
    /* d_in[2] enc_Whh unused: h0 == 0 */
    const float* enc_bih = (const float*)d_in[3];   // 2048
    const float* enc_bhh = (const float*)d_in[4];   // 2048
    const float* dec_Wih = (const float*)d_in[5];   // 2048 x 360
    const float* dec_Whh = (const float*)d_in[6];   // 2048 x 512
    const float* dec_bih = (const float*)d_in[7];   // 2048
    const float* dec_bhh = (const float*)d_in[8];   // 2048
    const float* fc_W    = (const float*)d_in[9];   // 360 x 512
    const float* fc_b    = (const float*)d_in[10];  // 360
    float* out = (float*)d_out;                     // 512 x 360 x 360 f32
    (void)in_sizes; (void)n_in; (void)out_size;

    char* ws = (char*)d_ws;
    size_t off = 0;
    auto alloc = [&](size_t bytes) -> void* {
        void* p = ws + off;
        off = (off + bytes + 255) & ~(size_t)255;
        return p;
    };
    bf16_t* xb       = (bf16_t*)alloc((size_t)BATCH * LPAD * 2);
    bf16_t* encWih_b = (bf16_t*)alloc((size_t)G4H * LPAD * 2);
    bf16_t* decWih_b = (bf16_t*)alloc((size_t)G4H * LPAD * 2);
    bf16_t* decWhh_b = (bf16_t*)alloc((size_t)G4H * HID * 2);
    bf16_t* fcW_b    = (bf16_t*)alloc((size_t)NPAD * HID * 2);
    float*  xproj_e  = (float*) alloc((size_t)BATCH * G4H * 4);
    float*  xproj_d  = (float*) alloc((size_t)BATCH * G4H * 4);
    float*  cbuf     = (float*) alloc((size_t)BATCH * HID * 4);
    bf16_t* hbuf0    = (bf16_t*)alloc((size_t)BATCH * HID * 2);
    bf16_t* hbuf1    = (bf16_t*)alloc((size_t)BATCH * HID * 2);
    if (off > ws_size) return;  // workspace too small: bail (no-op)

    // ---- prep: bf16 conversions with padding --------------------------------
    auto cvt = [&](const float* s, bf16_t* d, int sr, int sc, int dr, int dc) {
        int total = dr * dc;
        convert_pad_kernel<<<(total + 255) / 256, 256, 0, stream>>>(s, d, sr, sc, dr, dc);
    };
    cvt(x,       xb,       BATCH, SEQL, BATCH, LPAD);
    cvt(enc_Wih, encWih_b, G4H,   SEQL, G4H,   LPAD);
    cvt(dec_Wih, decWih_b, G4H,   SEQL, G4H,   LPAD);
    cvt(dec_Whh, decWhh_b, G4H,   HID,  G4H,   HID);
    cvt(fc_W,    fcW_b,    SEQL,  HID,  NPAD,  HID);

    // ---- input projections (K = 384, N = 2048, M = 512) ---------------------
    {
        int mt = BATCH / 16, nt = G4H / 16, kt = LPAD / 32;   // 32 x 128 tiles
        int blocks = (mt * nt) / 8;                            // 8 waves/block
        gemm_bf16_kernel<<<blocks, 256, 0, stream>>>(xb, LPAD, encWih_b, LPAD,
                                                     xproj_e, G4H, nt, kt);
        gemm_bf16_kernel<<<blocks, 256, 0, stream>>>(xb, LPAD, decWih_b, LPAD,
                                                     xproj_d, G4H, nt, kt);
    }
    // fold decoder biases into xproj_d once (step-invariant)
    addbias2_kernel<<<(BATCH * G4H + 255) / 256, 256, 0, stream>>>(
        xproj_d, dec_bih, dec_bhh, BATCH, G4H);

    // ---- encoder cell -> h0 (bf16), c (f32) ---------------------------------
    enc_cell_kernel<<<(BATCH * HID + 255) / 256, 256, 0, stream>>>(
        xproj_e, enc_bih, enc_bhh, cbuf, hbuf0);

    // ---- 360 fused decoder steps (h ping-pong); step t also emits pred_{t-1}
    const bf16_t* hin = hbuf0;
    bf16_t* hout = hbuf1;
    for (int t = 0; t < SEQL; ++t) {
        float* out_prev = (t == 0) ? nullptr : out + (size_t)(t - 1) * SEQL;
        step_fused_kernel<<<224, 256, 0, stream>>>(hin, hout, decWhh_b, xproj_d,
                                                   cbuf, fcW_b, fc_b, out_prev);
        const bf16_t* tmp = hout;
        hout = (bf16_t*)hin;
        hin = tmp;
    }
    // final projection: pred_359 = fc(h^360)  (h^360 is in `hin` after swaps)
    pred_kernel<<<92, 256, 0, stream>>>(hin, fcW_b, fc_b,
                                        out + (size_t)(SEQL - 1) * SEQL);
}
